// CrossLingualAlignmentHead_77094662963653
// MI455X (gfx1250) — matrix operand order repaired and checked by reading, not compiled, not run on hardware
//
#include <hip/hip_runtime.h>
#include <hip/hip_bf16.h>
#include <math.h>

typedef _Float16 f16;
typedef __attribute__((ext_vector_type(16))) _Float16 v16h;
typedef __attribute__((ext_vector_type(8)))  _Float16 v8h;
typedef __attribute__((ext_vector_type(8)))  float    v8f;

#define DAL  256
#define DSRC 512
#define ROWS 1024   // B*S = B*T = 4*256

// ---------------- prep kernels ----------------

// plain f32 -> f16 convert (source/target activations, row-major kept)
__global__ void cvt_f32_f16(const float* __restrict__ in, f16* __restrict__ out, int n) {
    int i = blockIdx.x * blockDim.x + threadIdx.x;
    if (i < n) out[i] = (f16)in[i];
}

// Pack a [K x 256] f32 weight matrix into per-lane WMMA B-fragment order:
//   p = (((kt*16 + nt)*32 + lane)*16 + e)
//   k = kt*32 + (lane>>4)*16 + e ; n = nt*16 + (lane&15)
// so each lane's 16-half B fragment is one contiguous 32B chunk.
__global__ void pack_w(const float* __restrict__ W, f16* __restrict__ out, int K) {
    int p = blockIdx.x * blockDim.x + threadIdx.x;
    if (p >= K * 256) return;
    int e  = p & 15;
    int L  = (p >> 4) & 31;
    int nt = (p >> 9) & 15;
    int kt = p >> 13;
    int k = kt * 32 + (L >> 4) * 16 + e;
    int n = nt * 16 + (L & 15);
    out[p] = (f16)W[k * 256 + n];
}

// ---------------- fused projection + scorer-input GEMMs ----------------
// blockIdx: bit0 selects source/target path, rest selects a 16-row strip.
// Phase 1: proj = act @ W + bias      (K=512), f32 -> d_out, f16 -> LDS
// Phase 2: h    = proj @ W1-half      (K=256), A-fragments read from LDS
__global__ __launch_bounds__(256)
void fused_proj(const f16* __restrict__ srch, const f16* __restrict__ tgth,
                const f16* __restrict__ Wsh,  const f16* __restrict__ Wth,
                const f16* __restrict__ W1h,
                const float* __restrict__ bs, const float* __restrict__ bt,
                float* __restrict__ sp_out, float* __restrict__ tp_out,
                float* __restrict__ hs_out, float* __restrict__ ht_out)
{
    __shared__ f16 pLds[16 * DAL];   // 8KB: f16 projection strip (bias included)

    const int mat   = blockIdx.x & 1;
    const int strip = blockIdx.x >> 1;       // 0..63
    const int lane  = threadIdx.x & 31;
    const int wave  = threadIdx.x >> 5;      // 0..7
    const int m     = lane & 15;
    const int hf    = lane >> 4;

    const f16*   A    = mat ? tgth   : srch;
    const f16*   Bw   = mat ? Wth    : Wsh;
    const float* bias = mat ? bt     : bs;
    float*       out1 = mat ? tp_out : sp_out;
    float*       out2 = mat ? ht_out : hs_out;
    const f16*   B2   = W1h + mat * 65536;   // W1 rows [0,256) -> hs, [256,512) -> ht

    const int row0 = strip * 16;

    // ---- Phase 1: K = 512 ----
    #pragma unroll
    for (int ci = 0; ci < 2; ++ci) {
        const int nt = wave * 2 + ci;        // column tile 0..15
        v8f acc = {};
        const f16* arow = A + (row0 + m) * DSRC;
        for (int kb = 0; kb < DSRC; kb += 32) {
            const f16* ap = arow + kb + hf * 16;
            v8h alo = *(const v8h*)(ap);
            v8h ahi = *(const v8h*)(ap + 8);
            v16h a;
            #pragma unroll
            for (int i = 0; i < 8; ++i) { a[i] = alo[i]; a[8 + i] = ahi[i]; }
            v16h b = *(const v16h*)(Bw + ((((kb >> 5) * 16 + nt) * 32 + lane) << 4));
            acc = __builtin_amdgcn_wmma_f32_16x16x32_f16(
                      false, a, false, b, (short)0, acc, false, false);
        }
        const int n  = nt * 16 + m;
        const float bn = bias[n];
        #pragma unroll
        for (int r = 0; r < 8; ++r) {
            const int rr = r + 8 * hf;
            const float v = acc[r] + bn;
            out1[(row0 + rr) * DAL + n] = v;
            pLds[rr * DAL + n] = (f16)v;
        }
    }
    __syncthreads();

    // ---- Phase 2: K = 256, A from LDS ----
    #pragma unroll
    for (int ci = 0; ci < 2; ++ci) {
        const int nt = wave * 2 + ci;
        v8f acc = {};
        for (int kb = 0; kb < DAL; kb += 32) {
            const f16* ap = pLds + m * DAL + kb + hf * 16;
            v8h alo = *(const v8h*)(ap);
            v8h ahi = *(const v8h*)(ap + 8);
            v16h a;
            #pragma unroll
            for (int i = 0; i < 8; ++i) { a[i] = alo[i]; a[8 + i] = ahi[i]; }
            v16h b = *(const v16h*)(B2 + ((((kb >> 5) * 16 + nt) * 32 + lane) << 4));
            acc = __builtin_amdgcn_wmma_f32_16x16x32_f16(
                      false, a, false, b, (short)0, acc, false, false);
        }
        const int n = nt * 16 + m;
        #pragma unroll
        for (int r = 0; r < 8; ++r)
            out2[(row0 + r + 8 * hf) * DAL + n] = acc[r];
    }
}

// ---------------- pairwise scorer (VALU; ReLU breaks matmul structure) ----------------
// block = (b, s); thread t owns scores[b,s,t]; hs-row + b1 and W2 staged in LDS.
__global__ __launch_bounds__(256)
void scorer(const float* __restrict__ hs, const float* __restrict__ ht,
            const float* __restrict__ b1, const float* __restrict__ W2,
            const float* __restrict__ b2, float* __restrict__ scores)
{
    __shared__ float hrow[DAL];
    __shared__ float w2[DAL];
    const int bid = blockIdx.x;          // b*256 + s
    const int t   = threadIdx.x;
    hrow[t] = hs[bid * DAL + t] + b1[t];
    w2[t]   = W2[t];
    __syncthreads();

    const int b = bid >> 8;
    const float* hr = ht + (b * 256 + t) * DAL;
    float acc = 0.f;
    #pragma unroll 4
    for (int h = 0; h < DAL; h += 4) {
        float4 hv = *(const float4*)(hr + h);
        acc += fmaxf(hrow[h]     + hv.x, 0.f) * w2[h];
        acc += fmaxf(hrow[h + 1] + hv.y, 0.f) * w2[h + 1];
        acc += fmaxf(hrow[h + 2] + hv.z, 0.f) * w2[h + 2];
        acc += fmaxf(hrow[h + 3] + hv.w, 0.f) * w2[h + 3];
    }
    const float x = acc + b2[0];
    scores[bid * 256 + t] = 1.0f / (1.0f + __expf(-x));
}

// ---------------- launcher ----------------
extern "C" void kernel_launch(void* const* d_in, const int* in_sizes, int n_in,
                              void* d_out, int out_size, void* d_ws, size_t ws_size,
                              hipStream_t stream) {
    const float* src = (const float*)d_in[0];
    const float* tgt = (const float*)d_in[1];
    const float* Ws  = (const float*)d_in[2];
    const float* bs  = (const float*)d_in[3];
    const float* Wt  = (const float*)d_in[4];
    const float* bt  = (const float*)d_in[5];
    const float* W1  = (const float*)d_in[6];
    const float* b1  = (const float*)d_in[7];
    const float* W2  = (const float*)d_in[8];
    const float* b2  = (const float*)d_in[9];
    (void)in_sizes; (void)n_in; (void)out_size; (void)ws_size;

    char* ws = (char*)d_ws;
    f16*   srch = (f16*)(ws);                    // 1,048,576 B
    f16*   tgth = (f16*)(ws + 1048576);          // 1,048,576 B
    f16*   Wsh  = (f16*)(ws + 2097152);          //   262,144 B
    f16*   Wth  = (f16*)(ws + 2359296);          //   262,144 B
    f16*   W1h  = (f16*)(ws + 2621440);          //   262,144 B
    float* hs   = (float*)(ws + 2883584);        // 1,048,576 B
    float* ht   = (float*)(ws + 3932160);        // 1,048,576 B

    float* scores = (float*)d_out;               // [4,256,256]
    float* sp     = scores + 4 * 256 * 256;      // [4,256,256]
    float* tp     = sp     + 4 * 256 * 256;      // [4,256,256]

    cvt_f32_f16<<<524288 / 256, 256, 0, stream>>>(src, srch, 524288);
    cvt_f32_f16<<<524288 / 256, 256, 0, stream>>>(tgt, tgth, 524288);
    pack_w<<<131072 / 256, 256, 0, stream>>>(Ws, Wsh, 512);
    pack_w<<<131072 / 256, 256, 0, stream>>>(Wt, Wth, 512);
    pack_w<<<131072 / 256, 256, 0, stream>>>(W1, W1h, 512);

    fused_proj<<<128, 256, 0, stream>>>(srch, tgth, Wsh, Wth, W1h,
                                        bs, bt, sp, tp, hs, ht);
    scorer<<<1024, 256, 0, stream>>>(hs, ht, b1, W2, b2, scores);
}
